// GraphTransformer_146028888376
// MI455X (gfx1250) — matrix operand order repaired and checked by reading, not compiled, not run on hardware
//
#include <hip/hip_runtime.h>
#include <stdint.h>

// ---------------- types / constants ----------------
typedef __bf16 bf16_t;
typedef __attribute__((ext_vector_type(16))) __bf16 v16bf;
typedef __attribute__((ext_vector_type(8)))  float  v8f;

// Exact pointee type expected by the async-LDS builtin per hipcc diagnostics:
//   "__attribute__((__vector_size__(4 * sizeof(int)))) int __device__ *"
typedef int v4i_vs __attribute__((vector_size(16)));
typedef __attribute__((address_space(1))) v4i_vs* gv4i_p;   // global (AS1)
typedef __attribute__((address_space(3))) v4i_vs* lv4i_p;   // LDS (AS3)

#define NN      20000
#define EE      320000
#define DMODEL  256
#define NHEAD   8
#define DFFN    1024
#define NLAYER  4

#define BM 128
#define BN 128
#define BK 32
#define GT 256   // GEMM threads per block (8 waves)

#if defined(__gfx1250__) && __has_builtin(__builtin_amdgcn_global_load_async_to_lds_b128) && __has_builtin(__builtin_amdgcn_s_wait_asynccnt)
#define USE_ASYNC_LDS 1
#else
#define USE_ASYNC_LDS 0
#endif

// 16B global -> LDS copy (async on CDNA5, sync fallback otherwise)
__device__ __forceinline__ void cp16_lds(void* lds, const void* g) {
#if USE_ASYNC_LDS
  // flat LDS address: low 32 bits are the LDS offset (ISA 10.2 aperture rules)
  __builtin_amdgcn_global_load_async_to_lds_b128(
      (gv4i_p)(uintptr_t)g,
      (lv4i_p)(unsigned)(uintptr_t)lds, 0, 0);
#else
  *(int4*)lds = *(const int4*)g;
#endif
}

template<int N>
__device__ __forceinline__ void cp_wait() {
#if USE_ASYNC_LDS
  __builtin_amdgcn_s_wait_asynccnt(N);
#endif
}

// ---------------- WMMA fragment loaders (ISA 7.12.2 16-bit layouts) ----------------
// A: 16x32 bf16, row-major tile in LDS, row stride BK elements.
// lane 0-15: M=lane, K 0..7/16..23 ; lane 16-31: M=lane-16, K offset +8/+24.
__device__ __forceinline__ v16bf frag_a(const bf16_t* base, int lane) {
  const int r = lane & 15, h = lane >> 4;
  union { v16bf v; unsigned u[8]; } f;
  const unsigned* rp = (const unsigned*)(base + r * BK);
#pragma unroll
  for (int i = 0; i < 8; ++i)
    f.u[i] = rp[((i < 4) ? 0 : 8) + (i & 3) + h * 4];   // K pair (v<4?0:16)+2(v&3)+8h
  return f.v;
}
// B: 32x16 bf16; we store B transposed ([N][K]) so B[k][n],B[k+1][n] are contiguous.
// lane half h holds K 8h..8h+7 and 16+8h..16+8h+7; col = lane&15.
__device__ __forceinline__ v16bf frag_b(const bf16_t* base, int lane) {
  const int n = lane & 15, h = lane >> 4;
  union { v16bf v; unsigned u[8]; } f;
  const unsigned* rp = (const unsigned*)(base + n * BK);
#pragma unroll
  for (int i = 0; i < 8; ++i)
    f.u[i] = rp[((i < 4) ? 0 : 8) + (i & 3) + h * 4];   // K pair (v<4?0:16)+2(v&3)+8h
  return f.v;
}

// ---------------- GEMM: C[M,N] = A_bf16[M,K] @ Bt_bf16[N,K]^T + bias ----------------
// MODE 0: f32 output + bias ; MODE 1: bf16 output + bias + ReLU
template<int MODE>
__global__ __launch_bounds__(GT)
void k_gemm(const bf16_t* __restrict__ A, const bf16_t* __restrict__ Bt,
            const float* __restrict__ bias,
            float* __restrict__ outF, bf16_t* __restrict__ outB,
            int M, int N, int K)
{
  __shared__ __align__(16) bf16_t sA[2][BM * BK];
  __shared__ __align__(16) bf16_t sB[2][BN * BK];

  const int tid  = threadIdx.x;
  const int lane = tid & 31;
  const int wid  = tid >> 5;
  const int wm   = wid >> 2;           // 0..1 : 64 rows per wave
  const int wn   = wid & 3;            // 0..3 : 32 cols per wave
  const int n0   = blockIdx.x * BN;
  const int m0   = blockIdx.y * BM;

  v8f zero;
#pragma unroll
  for (int r = 0; r < 8; ++r) zero[r] = 0.0f;
  v8f acc[4][2];
#pragma unroll
  for (int i = 0; i < 4; ++i)
#pragma unroll
    for (int j = 0; j < 2; ++j) acc[i][j] = zero;

  // Stage copy: 4 async b128 per thread per stage (uniform -> asynccnt bookkeeping).
  auto stage_copy = [&](int st, int k0) {
#pragma unroll
    for (int i = 0; i < 2; ++i) {                    // A tile: 128 rows x 64B
      int c = tid + i * GT;                          // 0..511
      int row = c >> 2, ce = (c & 3) * 8;
      int ga = m0 + row; ga = (ga < M) ? ga : (M - 1);   // clamp tail (rows unused)
      cp16_lds(&sA[st][row * BK + ce], A + (size_t)ga * K + k0 + ce);
    }
#pragma unroll
    for (int i = 0; i < 2; ++i) {                    // B tile (N always %128==0)
      int c = tid + i * GT;
      int row = c >> 2, ce = (c & 3) * 8;
      cp16_lds(&sB[st][row * BK + ce], Bt + (size_t)(n0 + row) * K + k0 + ce);
    }
  };

  const int S = K / BK;
  stage_copy(0, 0);
  for (int ks = 0; ks < S; ++ks) {
    const int st = ks & 1;
    if (ks + 1 < S) { stage_copy(st ^ 1, (ks + 1) * BK); cp_wait<4>(); }
    else            { cp_wait<0>(); }
    __syncthreads();                                   // current stage resident

    const bf16_t* At = sA[st];
    const bf16_t* Bl = sB[st];
    v16bf a[4], b[2];
#pragma unroll
    for (int i = 0; i < 4; ++i) a[i] = frag_a(At + (wm * 64 + i * 16) * BK, lane);
#pragma unroll
    for (int j = 0; j < 2; ++j) b[j] = frag_b(Bl + (wn * 32 + j * 16) * BK, lane);
#pragma unroll
    for (int i = 0; i < 4; ++i)
#pragma unroll
      for (int j = 0; j < 2; ++j)
        acc[i][j] = __builtin_amdgcn_wmma_f32_16x16x32_bf16(
            false, a[i], false, b[j], (short)0, acc[i][j], false, false);
    __syncthreads();                                   // done reading before re-fill
  }

  // Epilogue per C/D layout: VGPR r -> M = r + 8*(lane>=16), N = lane&15.
  const int half = lane >> 4;
  const int cn   = lane & 15;
#pragma unroll
  for (int i = 0; i < 4; ++i) {
    const int row0 = m0 + wm * 64 + i * 16 + half * 8;
#pragma unroll
    for (int j = 0; j < 2; ++j) {
      const int col = n0 + wn * 32 + j * 16 + cn;
      const float bv = bias[col];
#pragma unroll
      for (int r = 0; r < 8; ++r) {
        const int row = row0 + r;
        if (row < M) {
          float v = acc[i][j][r] + bv;
          if (MODE == 1) outB[(size_t)row * N + col] = (bf16_t)fmaxf(v, 0.0f);
          else           outF[(size_t)row * N + col] = v;
        }
      }
    }
  }
}

// ---------------- small utility kernels ----------------
__global__ void k_tobf16(const float* __restrict__ in, bf16_t* __restrict__ out, int n) {
  int i = blockIdx.x * blockDim.x + threadIdx.x;
  if (i < n) out[i] = (bf16_t)in[i];
}

// W[K][N] f32 -> Wt[N][K] bf16
__global__ void k_transpose_bf16(const float* __restrict__ W, bf16_t* __restrict__ Wt,
                                 int K, int N) {
  int i = blockIdx.x * blockDim.x + threadIdx.x;
  if (i < K * N) {
    int k = i / N, n = i - k * N;
    Wt[(size_t)n * K + k] = (bf16_t)W[(size_t)k * N + n];
  }
}

__global__ void k_add(float* __restrict__ y, const float* __restrict__ a, int n) {
  int i = blockIdx.x * blockDim.x + threadIdx.x;
  if (i < n) y[i] += a[i];
}

__global__ void k_concat_bf16(const float* __restrict__ x, const float* __restrict__ ip,
                              const float* __restrict__ ep, bf16_t* __restrict__ out, int n) {
  int i = blockIdx.x * blockDim.x + threadIdx.x;
  if (i < n) {
    int row = i / (3 * DMODEL), c = i - row * 3 * DMODEL;
    float v = (c < DMODEL) ? x[(size_t)row * DMODEL + c]
            : (c < 2 * DMODEL) ? ip[(size_t)row * DMODEL + c - DMODEL]
                               : ep[(size_t)row * DMODEL + c - 2 * DMODEL];
    out[i] = (bf16_t)v;
  }
}

__global__ void k_seg_init(float* smax, float* ssum, int n) {
  int i = blockIdx.x * blockDim.x + threadIdx.x;
  if (i < n) { smax[i] = -__builtin_inff(); ssum[i] = 0.0f; }
}

// wave-per-row LayerNorm over DMODEL; writes f32 and optional bf16
__global__ void k_ln(const float* __restrict__ x, const float* __restrict__ g,
                     const float* __restrict__ b, float* __restrict__ y,
                     bf16_t* __restrict__ ybf, int rows) {
  int row = (int)((blockIdx.x * blockDim.x + threadIdx.x) >> 5);
  int lane = threadIdx.x & 31;
  if (row >= rows) return;
  const float* xr = x + (size_t)row * DMODEL;
  float v[8], s = 0.0f;
#pragma unroll
  for (int i = 0; i < 8; ++i) { v[i] = xr[lane + i * 32]; s += v[i]; }
#pragma unroll
  for (int o = 16; o > 0; o >>= 1) s += __shfl_xor(s, o);
  float mu = s * (1.0f / DMODEL);
  float var = 0.0f;
#pragma unroll
  for (int i = 0; i < 8; ++i) { float d = v[i] - mu; var += d * d; }
#pragma unroll
  for (int o = 16; o > 0; o >>= 1) var += __shfl_xor(var, o);
  float rs = rsqrtf(var * (1.0f / DMODEL) + 1e-5f);
#pragma unroll
  for (int i = 0; i < 8; ++i) {
    int c = lane + i * 32;
    float o = (v[i] - mu) * rs * g[c] + b[c];
    y[(size_t)row * DMODEL + c] = o;
    if (ybf) ybf[(size_t)row * DMODEL + c] = (bf16_t)o;
  }
}

__device__ __forceinline__ void atomicMaxF(float* addr, float val) {
  unsigned* a = (unsigned*)addr;
  unsigned old = *a;
  while (true) {
    float f = __uint_as_float(old);
    if (f >= val) break;
    unsigned assumed = old;
    old = atomicCAS(a, assumed, __float_as_uint(val));
    if (old == assumed) break;
  }
}

// ---------------- edge kernels (wave-per-edge; lane -> 8 channels, head = lane>>2) ----------------
__global__ void k_edge_alpha(const float* __restrict__ q, const float* __restrict__ k,
                             const float* __restrict__ ea, const float* __restrict__ We,
                             const int* __restrict__ src, const int* __restrict__ dst,
                             float* __restrict__ alpha, float* __restrict__ smax, int E) {
  int e = (int)((blockIdx.x * blockDim.x + threadIdx.x) >> 5);
  int lane = threadIdx.x & 31;
  if (e >= E) return;
  int s = src[e], d = dst[e];
  float ar[16];
#pragma unroll
  for (int j = 0; j < 16; ++j) ar[j] = ea[(size_t)e * 16 + j];
  float acc = 0.0f;
#pragma unroll
  for (int i = 0; i < 8; ++i) {
    int ch = lane * 8 + i;
    float ev = 0.0f;
#pragma unroll
    for (int j = 0; j < 16; ++j) ev += ar[j] * We[j * DMODEL + ch];
    acc += q[(size_t)d * DMODEL + ch] * (k[(size_t)s * DMODEL + ch] + ev);
  }
  acc += __shfl_xor(acc, 1);
  acc += __shfl_xor(acc, 2);          // reduce over the 4 lanes of a head
  if ((lane & 3) == 0) {
    int h = lane >> 2;
    float a = acc * 0.17677669529663687f;   // 1/sqrt(32)
    alpha[(size_t)e * NHEAD + h] = a;
    atomicMaxF(&smax[(size_t)d * NHEAD + h], a);
  }
}

__global__ void k_edge_expsum(float* __restrict__ alpha, const float* __restrict__ smax,
                              float* __restrict__ ssum, const int* __restrict__ dst, int n) {
  int i = blockIdx.x * blockDim.x + threadIdx.x;
  if (i >= n) return;
  int e = i >> 3, h = i & 7;
  int d = dst[e];
  float ex = __expf(alpha[i] - smax[(size_t)d * NHEAD + h]);
  alpha[i] = ex;
  atomicAdd(&ssum[(size_t)d * NHEAD + h], ex);
}

__global__ void k_edge_msg(const float* __restrict__ v, const float* __restrict__ ea,
                           const float* __restrict__ We, const int* __restrict__ src,
                           const int* __restrict__ dst, const float* __restrict__ ex,
                           const float* __restrict__ ssum, float* __restrict__ out, int E) {
  int e = (int)((blockIdx.x * blockDim.x + threadIdx.x) >> 5);
  int lane = threadIdx.x & 31;
  if (e >= E) return;
  int s = src[e], d = dst[e], h = lane >> 2;
  float w = ex[(size_t)e * NHEAD + h] / (ssum[(size_t)d * NHEAD + h] + 1e-16f);
  float ar[16];
#pragma unroll
  for (int j = 0; j < 16; ++j) ar[j] = ea[(size_t)e * 16 + j];
#pragma unroll
  for (int i = 0; i < 8; ++i) {
    int ch = lane * 8 + i;
    float ev = 0.0f;
#pragma unroll
    for (int j = 0; j < 16; ++j) ev += ar[j] * We[j * DMODEL + ch];
    atomicAdd(&out[(size_t)d * DMODEL + ch], (v[(size_t)s * DMODEL + ch] + ev) * w);
  }
}

// ---------------- host orchestration ----------------
extern "C" void kernel_launch(void* const* d_in, const int* in_sizes, int n_in,
                              void* d_out, int out_size, void* d_ws, size_t ws_size,
                              hipStream_t stream) {
  (void)in_sizes; (void)n_in; (void)out_size; (void)ws_size;
  const float* input = (const float*)d_in[0];
  const float* nemb  = (const float*)d_in[1];
  const float* eattr = (const float*)d_in[2];
  const int*   eidx  = (const int*)d_in[3];
  const float* Win   = (const float*)d_in[4];  const float* b_in = (const float*)d_in[5];
  const float* Wemb  = (const float*)d_in[6];  const float* bemb = (const float*)d_in[7];
  const float* Wfus  = (const float*)d_in[8];  const float* bfus = (const float*)d_in[9];
  const float* Wq    = (const float*)d_in[10]; const float* bq   = (const float*)d_in[11];
  const float* Wk    = (const float*)d_in[12]; const float* bk   = (const float*)d_in[13];
  const float* Wv    = (const float*)d_in[14]; const float* bv   = (const float*)d_in[15];
  const float* We    = (const float*)d_in[16];
  const float* Wsk   = (const float*)d_in[17]; const float* bsk  = (const float*)d_in[18];
  const float* n1g   = (const float*)d_in[19]; const float* n1b  = (const float*)d_in[20];
  const float* W1f   = (const float*)d_in[21]; const float* b1f  = (const float*)d_in[22];
  const float* W2f   = (const float*)d_in[23]; const float* b2f  = (const float*)d_in[24];
  const float* n2g   = (const float*)d_in[25]; const float* n2b  = (const float*)d_in[26];
  const float* fng   = (const float*)d_in[27]; const float* fnb  = (const float*)d_in[28];
  const int* src = eidx;
  const int* dst = eidx + EE;

  char* wp = (char*)d_ws;
  auto alloc = [&](size_t bytes) -> void* {
    void* p = wp; wp += (bytes + 255) & ~(size_t)255; return p;
  };
  const size_t ND = (size_t)NN * DMODEL;
  float* inp_proj = (float*)alloc(ND * 4);
  float* emb_proj = (float*)alloc(ND * 4);
  float* xbuf     = (float*)alloc(ND * 4);
  float* xn       = (float*)alloc(ND * 4);
  float* qb       = (float*)alloc(ND * 4);
  float* kb       = (float*)alloc(ND * 4);
  float* vb       = (float*)alloc(ND * 4);
  float* attn     = (float*)alloc(ND * 4);
  float* tmp      = (float*)alloc(ND * 4);
  bf16_t* xn_bf   = (bf16_t*)alloc(ND * 2);
  bf16_t* comb_bf = (bf16_t*)alloc((size_t)NN * 3 * DMODEL * 2);
  bf16_t* h_bf    = (bf16_t*)alloc((size_t)NN * DFFN * 2);
  bf16_t* in_bf   = (bf16_t*)alloc((size_t)NN * 64 * 2);
  bf16_t* em_bf   = (bf16_t*)alloc((size_t)NN * 64 * 2);
  float* alpha    = (float*)alloc((size_t)EE * NHEAD * 4);
  float* smax     = (float*)alloc((size_t)NN * NHEAD * 4);
  float* ssum     = (float*)alloc((size_t)NN * NHEAD * 4);
  bf16_t* WinT    = (bf16_t*)alloc((size_t)64 * DMODEL * 2);
  bf16_t* WembT   = (bf16_t*)alloc((size_t)64 * DMODEL * 2);
  bf16_t* WfusT   = (bf16_t*)alloc((size_t)3 * DMODEL * DMODEL * 2);
  bf16_t* WqT     = (bf16_t*)alloc((size_t)NLAYER * DMODEL * DMODEL * 2);
  bf16_t* WkT     = (bf16_t*)alloc((size_t)NLAYER * DMODEL * DMODEL * 2);
  bf16_t* WvT     = (bf16_t*)alloc((size_t)NLAYER * DMODEL * DMODEL * 2);
  bf16_t* WskT    = (bf16_t*)alloc((size_t)NLAYER * DMODEL * DMODEL * 2);
  bf16_t* W1T     = (bf16_t*)alloc((size_t)NLAYER * DMODEL * DFFN * 2);
  bf16_t* W2T     = (bf16_t*)alloc((size_t)NLAYER * DFFN * DMODEL * 2);

  auto cdiv = [](int a, int b) { return (a + b - 1) / b; };
  auto transpose = [&](const float* W, bf16_t* Wt, int K, int N) {
    k_transpose_bf16<<<cdiv(K * N, 256), 256, 0, stream>>>(W, Wt, K, N);
  };
  auto gemmF = [&](const bf16_t* A, const bf16_t* Bt, const float* bias,
                   float* out, int M, int N, int K) {
    dim3 g(N / BN, cdiv(M, BM));
    k_gemm<0><<<g, GT, 0, stream>>>(A, Bt, bias, out, nullptr, M, N, K);
  };
  auto gemmR = [&](const bf16_t* A, const bf16_t* Bt, const float* bias,
                   bf16_t* out, int M, int N, int K) {
    dim3 g(N / BN, cdiv(M, BM));
    k_gemm<1><<<g, GT, 0, stream>>>(A, Bt, bias, nullptr, out, M, N, K);
  };

  // ---- weight prep (bf16, transposed to [N][K]) ----
  transpose(Win,  WinT,  64, DMODEL);
  transpose(Wemb, WembT, 64, DMODEL);
  transpose(Wfus, WfusT, 3 * DMODEL, DMODEL);
  for (int l = 0; l < NLAYER; ++l) {
    size_t o = (size_t)l * DMODEL * DMODEL, of = (size_t)l * DMODEL * DFFN;
    transpose(Wq  + o,  WqT  + o,  DMODEL, DMODEL);
    transpose(Wk  + o,  WkT  + o,  DMODEL, DMODEL);
    transpose(Wv  + o,  WvT  + o,  DMODEL, DMODEL);
    transpose(Wsk + o,  WskT + o,  DMODEL, DMODEL);
    transpose(W1f + of, W1T  + of, DMODEL, DFFN);
    transpose(W2f + of, W2T  + of, DFFN, DMODEL);
  }

  // ---- input projections ----
  k_tobf16<<<cdiv(NN * 64, 256), 256, 0, stream>>>(input, in_bf, NN * 64);
  k_tobf16<<<cdiv(NN * 64, 256), 256, 0, stream>>>(nemb,  em_bf, NN * 64);
  gemmF(in_bf, WinT,  b_in, inp_proj, NN, DMODEL, 64);
  gemmF(em_bf, WembT, bemb, emb_proj, NN, DMODEL, 64);

  const float* xcur = emb_proj;   // x starts as emb_proj
  const int lnBlocks = cdiv(NN, 8);
  const int edgeBlocks = cdiv(EE, 8);

  for (int l = 0; l < NLAYER; ++l) {
    size_t o = (size_t)l * DMODEL * DMODEL, of = (size_t)l * DMODEL * DFFN;
    // fusion: x = concat(x, inp_proj, emb_proj) @ Wfus + bfus
    k_concat_bf16<<<cdiv(NN * 3 * DMODEL, 256), 256, 0, stream>>>(xcur, inp_proj, emb_proj,
                                                                  comb_bf, NN * 3 * DMODEL);
    gemmF(comb_bf, WfusT, bfus, xbuf, NN, DMODEL, 3 * DMODEL);
    xcur = xbuf;

    // pre-norm attention
    k_ln<<<lnBlocks, 256, 0, stream>>>(xbuf, n1g + (size_t)l * DMODEL, n1b + (size_t)l * DMODEL,
                                       xn, xn_bf, NN);
    gemmF(xn_bf, WqT + o,  bq  + (size_t)l * DMODEL, qb,   NN, DMODEL, DMODEL);
    gemmF(xn_bf, WkT + o,  bk  + (size_t)l * DMODEL, kb,   NN, DMODEL, DMODEL);
    gemmF(xn_bf, WvT + o,  bv  + (size_t)l * DMODEL, vb,   NN, DMODEL, DMODEL);
    gemmF(xn_bf, WskT + o, bsk + (size_t)l * DMODEL, attn, NN, DMODEL, DMODEL);  // skip term

    k_seg_init<<<cdiv(NN * NHEAD, 256), 256, 0, stream>>>(smax, ssum, NN * NHEAD);
    const float* Wel = We + (size_t)l * 16 * DMODEL;
    k_edge_alpha<<<edgeBlocks, 256, 0, stream>>>(qb, kb, eattr, Wel, src, dst, alpha, smax, EE);
    k_edge_expsum<<<cdiv(EE * NHEAD, 256), 256, 0, stream>>>(alpha, smax, ssum, dst, EE * NHEAD);
    k_edge_msg<<<edgeBlocks, 256, 0, stream>>>(vb, eattr, Wel, src, dst, alpha, ssum, attn, EE);
    k_add<<<cdiv((int)ND, 256), 256, 0, stream>>>(xbuf, attn, (int)ND);

    // pre-norm FFN
    k_ln<<<lnBlocks, 256, 0, stream>>>(xbuf, n2g + (size_t)l * DMODEL, n2b + (size_t)l * DMODEL,
                                       xn, xn_bf, NN);
    gemmR(xn_bf, W1T + of, b1f + (size_t)l * DFFN, h_bf, NN, DFFN, DMODEL);   // ReLU -> bf16
    gemmF(h_bf, W2T + of, b2f + (size_t)l * DMODEL, tmp, NN, DMODEL, DFFN);
    k_add<<<cdiv((int)ND, 256), 256, 0, stream>>>(xbuf, tmp, (int)ND);
  }

  // final LayerNorm into d_out (f32)
  k_ln<<<lnBlocks, 256, 0, stream>>>(xbuf, fng, fnb, (float*)d_out, nullptr, NN);
}
